// Model_23965917512128
// MI455X (gfx1250) — matrix operand order repaired and checked by reading, not compiled, not run on hardware
//
#include <hip/hip_runtime.h>
#include <hip/hip_bf16.h>
#include <math.h>

// ---------------------------------------------------------------------------
// MoE pipeline for MI455X (gfx1250, wave32, WMMA).
//   B=4 T=24 H=32 W=32 D=128, 5x stride-2 2x2 convs (implicit GEMM K=512),
//   DFT gating -> top-2 experts, expert GEMMs K=128 N=128 via bf16 WMMA.
//   Expert weights staged to LDS via async DMA (global_load_async_to_lds).
// ---------------------------------------------------------------------------

typedef __attribute__((ext_vector_type(16))) __bf16 v16bf;
typedef __attribute__((ext_vector_type(8)))  float  v8f;

union V16 { v16bf v; uint4 q[2]; };

#define WMMA_BF16(a, b, c) \
  __builtin_amdgcn_wmma_f32_16x16x32_bf16(false, (a), false, (b), (short)0, (c), false, false)

__device__ __forceinline__ unsigned short f2bf(float f) {
  unsigned int u = __float_as_uint(f);
  u += 0x7fffu + ((u >> 16) & 1u);            // round-to-nearest-even
  return (unsigned short)(u >> 16);
}
__device__ __forceinline__ float bf2f(unsigned short h) {
  return __uint_as_float(((unsigned int)h) << 16);
}

// ------------------------------ constants ----------------------------------
#define BB 4
#define TT 24
#define HH 32
#define WW 32
#define DD 128
#define NIMG (BB * TT)              // 96
#define NEXP 7
#define TOKENS (BB * TT * HH * WW)  // 98304
#define TOK_PER_B (TT * HH * WW)    // 24576
#define CONV_K 512                  // 4 taps * 128 ch
#define COMBINE_EPS 2.220446049250313e-16f

// ---------------------------------------------------------------------------
// Kernel 0: weight prep.
//  - conv weights OIHW -> bf16 [l][o][k], k = tap*128 + ci
//  - expert weights edf -> bf16 [e][f][d]
//  - fold BN:  scale = gamma*rsqrt(var+eps); bias = scale*(conv_b-mean)+beta
// ---------------------------------------------------------------------------
__global__ void prep_kernel(const float* __restrict__ conv_w,
                            const float* __restrict__ conv_b,
                            const float* __restrict__ bn_gamma,
                            const float* __restrict__ bn_beta,
                            const float* __restrict__ bn_mean,
                            const float* __restrict__ bn_var,
                            const float* __restrict__ expert_w,
                            unsigned short* __restrict__ wT_conv,
                            unsigned short* __restrict__ wT_exp,
                            float* __restrict__ sb) {
  const int n = blockIdx.x * blockDim.x + threadIdx.x;
  const int stride = gridDim.x * blockDim.x;
  // conv: 5*128*512
  for (int i = n; i < 5 * 128 * CONV_K; i += stride) {
    int k   = i & 511;
    int tap = k >> 7;
    int ci  = k & 127;
    wT_conv[i] = f2bf(conv_w[(i & ~511) + ci * 4 + tap]);
  }
  // experts: 7*128*128, [e][f][d] <- [e][d][f]
  for (int i = n; i < NEXP * 128 * 128; i += stride) {
    int e = i >> 14;
    int r = i & 16383;
    int f = r >> 7;
    int d = r & 127;
    wT_exp[i] = f2bf(expert_w[(e << 14) + d * 128 + f]);
  }
  // BN fold: 5*128
  for (int i = n; i < 5 * 128; i += stride) {
    float sc = bn_gamma[i] * rsqrtf(bn_var[i] + 1e-5f);
    sb[i]       = sc;
    sb[640 + i] = sc * (conv_b[i] - bn_mean[i]) + bn_beta[i];
  }
}

// ---------------------------------------------------------------------------
// Kernel 1: one conv level as implicit GEMM.
//   M = NIMG*s_out^2 rows (output pixels), K = 512, N = 128.
//   Block = 128 threads (4 waves); block tile = 64 rows x 128 out-ch.
//   Patches staged in LDS (64 x 512 bf16 = 64 KB, dynamic LDS).
//   Fragment layouts per CDNA5 ISA 7.12.2 (16-bit A 16x32, B 32x16).
// ---------------------------------------------------------------------------
template <bool FIRST>
__global__ void conv_kernel(const float* __restrict__ xin,
                            const unsigned short* __restrict__ actIn,
                            const unsigned short* __restrict__ wT,    // [n=128][k=512] bf16
                            const float* __restrict__ scale,
                            const float* __restrict__ bias,
                            unsigned short* __restrict__ actOut,      // [img][oy][ox][c] bf16
                            int s_in, int s_out, int M) {
  extern __shared__ unsigned short patch[];  // [64][512]
  const int tid = threadIdx.x;
  const int so2 = s_out * s_out;

  // ---- cooperative patch gather: 64 rows x 4 taps, 128 ch each ----
  for (int p = tid; p < 256; p += 128) {
    int m = p >> 2, tap = p & 3;
    long gm = (long)blockIdx.x * 64 + m;
    if (gm >= M) gm = M - 1;                  // clamp (keeps EXEC full for WMMA)
    int img = (int)(gm / so2);
    int rem = (int)(gm % so2);
    int oy = rem / s_out, ox = rem % s_out;
    int iy = oy * 2 + (tap >> 1), ix = ox * 2 + (tap & 1);
    unsigned short* dst = &patch[m * CONV_K + tap * 128];
    if (FIRST) {
      const float* src = xin + (((long)img * HH + iy) * WW + ix) * DD;
      for (int c = 0; c < 128; c += 4) {
        float4 v = *(const float4*)(src + c);
        dst[c + 0] = f2bf(v.x); dst[c + 1] = f2bf(v.y);
        dst[c + 2] = f2bf(v.z); dst[c + 3] = f2bf(v.w);
      }
    } else {
      const unsigned short* src = actIn + (((long)img * s_in + iy) * s_in + ix) * DD;
      for (int c = 0; c < 128; c += 8)
        *(uint4*)(dst + c) = *(const uint4*)(src + c);
    }
  }
  __syncthreads();

  // ---- WMMA: each wave -> 16 rows x 128 out-ch (8 N-tiles) ----
  const int wv = tid >> 5, lane = tid & 31;
  const int lg = lane >> 4, lr = lane & 15;
  const int m0 = wv * 16;

  v8f zero = {};
  v8f acc[8];
#pragma unroll
  for (int nt = 0; nt < 8; ++nt) acc[nt] = zero;

  for (int kk = 0; kk < 16; ++kk) {          // K = 512 -> 16 steps of 32
    V16 a;
    const unsigned short* ap = &patch[(m0 + lr) * CONV_K + kk * 32 + lg * 8];
    a.q[0] = *(const uint4*)ap;              // K = base .. base+7
    a.q[1] = *(const uint4*)(ap + 16);       // K = base+16 .. base+23
#pragma unroll
    for (int nt = 0; nt < 8; ++nt) {
      V16 bfr;
      const unsigned short* bp = wT + (nt * 16 + lr) * CONV_K + kk * 32 + lg * 8;
      bfr.q[0] = *(const uint4*)bp;          // K = base .. base+7, base+16 .. base+23
      bfr.q[1] = *(const uint4*)(bp + 16);
      acc[nt] = WMMA_BF16(a.v, bfr.v, acc[nt]);
    }
  }

  // ---- epilogue: folded BN + leaky ReLU, store bf16 ----
  float sc[8], bi[8];
#pragma unroll
  for (int nt = 0; nt < 8; ++nt) {
    int c = nt * 16 + lr;
    sc[nt] = scale[c];
    bi[nt] = bias[c];
  }
#pragma unroll
  for (int nt = 0; nt < 8; ++nt) {
    int c = nt * 16 + lr;
#pragma unroll
    for (int r = 0; r < 8; ++r) {
      long gm = (long)blockIdx.x * 64 + m0 + r + lg * 8;  // C/D layout: M = r + 8*hi-half
      if (gm < M) {
        float v = acc[nt][r] * sc[nt] + bi[nt];
        v = v > 0.f ? v : 0.2f * v;
        actOut[gm * 128 + c] = f2bf(v);
      }
    }
  }
}

// ---------------------------------------------------------------------------
// Kernel 2: fuse -> 24-pt DFT (ortho) -> logits -> top-2 softmax gates.
// Single block, tiny.
// ---------------------------------------------------------------------------
__global__ void gate_kernel(const unsigned short* __restrict__ act4,  // [96][128] bf16
                            const float* __restrict__ fuse_w,
                            const float* __restrict__ fuse_b,
                            const float* __restrict__ w_gate,         // [12][7]
                            int* __restrict__ gidx,
                            float* __restrict__ gval) {
  __shared__ float s[NIMG];
  __shared__ float amp[BB][12];
  __shared__ float logits[BB][NEXP];
  const int tid = threadIdx.x;

  if (tid < NIMG) {
    float acc = 0.f;
    for (int c = 0; c < 128; ++c) acc += bf2f(act4[tid * 128 + c]) * fuse_w[c];
    s[tid] = acc + fuse_b[0];
  }
  __syncthreads();

  if (tid < BB * 12) {                        // freqs k = 1..12
    int b = tid / 12, k = tid % 12 + 1;
    float re = 0.f, im = 0.f;
    for (int t = 0; t < TT; ++t) {
      int ph = (k * t) % TT;                  // reduce phase mod 2*pi
      float a = -6.283185307179586f * (float)ph / (float)TT;
      re += s[b * TT + t] * __cosf(a);
      im += s[b * TT + t] * __sinf(a);
    }
    amp[b][k - 1] = sqrtf(re * re + im * im) * 0.20412414523193154f;  // 1/sqrt(24)
  }
  __syncthreads();

  if (tid < BB * NEXP) {
    int b = tid / NEXP, e = tid % NEXP;
    float acc = 0.f;
    for (int k = 0; k < 12; ++k) acc += amp[b][k] * w_gate[k * NEXP + e];
    logits[b][e] = acc;
  }
  __syncthreads();

  if (tid < BB) {
    int b = tid;
    int i1 = 0; float v1 = logits[b][0];
    for (int e = 1; e < NEXP; ++e)
      if (logits[b][e] > v1) { v1 = logits[b][e]; i1 = e; }
    int i2 = (i1 == 0) ? 1 : 0; float v2 = logits[b][i2];
    for (int e = 0; e < NEXP; ++e)
      if (e != i1 && logits[b][e] > v2) { v2 = logits[b][e]; i2 = e; }
    float e1 = 1.f, e2 = __expf(v2 - v1);     // softmax over top-2 (max = v1)
    float inv = 1.f / (e1 + e2);
    gidx[b * 2 + 0] = i1; gidx[b * 2 + 1] = i2;
    gval[b * 2 + 0] = e1 * inv; gval[b * 2 + 1] = e2 * inv;
  }
}

// ---------------------------------------------------------------------------
// Kernel 3: top-2 expert GEMMs + log(sum g*exp(.)).
//   Block = 256 threads (8 waves), tile = 128 tokens x 128 f.
//   LDS layout (dynamic, 96KB):
//     [0      , 32KB) : x tile, 128x128 bf16 (fp32->bf16 via VALU)
//     [32KB   , 64KB) : expert-0 weight tile [f][d] bf16 (async DMA)
//     [64KB   , 96KB) : expert-1 weight tile [f][d] bf16 (async DMA)
//   Weight tiles DMA'd once per block with global_load_async_to_lds_b128
//   (ASYNCcnt), overlapped with the VALU x-staging; 8 waves reuse them from
//   LDS instead of each re-reading 32KB from L2.
// ---------------------------------------------------------------------------
#define XS_USH (128 * 128)          // x tile ushorts (32KB)
#define WL_USH (128 * 128)          // per-expert weight ushorts (32KB)

__global__ void __launch_bounds__(256)
expert_kernel(const float* __restrict__ x,
              const unsigned short* __restrict__ wT_exp,   // [e][f][d] bf16
              const float* __restrict__ expert_b,          // [e][128]
              const int* __restrict__ gidx,
              const float* __restrict__ gval,
              float* __restrict__ out) {
  extern __shared__ unsigned short smem[];     // 96KB: xs | w0 | w1
  unsigned short* xs = smem;
  const int tid = threadIdx.x;
  const long row0 = (long)blockIdx.x * 128;
  const int b = (int)(row0 / TOK_PER_B);
  const int e0 = gidx[b * 2 + 0];
  const int e1 = gidx[b * 2 + 1];

  // ---- async DMA: both expert weight tiles (2 x 32KB bf16) -> LDS ----
  // 256 threads x 8 x b128 per expert = 32KB; LDS dest = byte offset
  // (dynamic LDS base is 0: no static LDS in this kernel).
  {
    const unsigned short* w0 = wT_exp + (long)e0 * WL_USH;
    const unsigned short* w1 = wT_exp + (long)e1 * WL_USH;
#pragma unroll
    for (int j = 0; j < 8; ++j) {
      unsigned int byteoff = (unsigned int)tid * 16u + (unsigned int)j * 4096u;
      unsigned int lds0 = (unsigned int)(XS_USH * 2) + byteoff;
      unsigned int lds1 = (unsigned int)(XS_USH * 2 + WL_USH * 2) + byteoff;
      asm volatile("global_load_async_to_lds_b128 %0, %1, %2 offset:0"
                   :
                   : "v"(lds0), "v"(byteoff), "s"(w0)
                   : "memory");
      asm volatile("global_load_async_to_lds_b128 %0, %1, %2 offset:0"
                   :
                   : "v"(lds1), "v"(byteoff), "s"(w1)
                   : "memory");
    }
  }

  // ---- stage x tile: 128 tokens x 128 d, fp32 -> bf16 (overlaps DMA) ----
  const float* src = x + row0 * DD;
  for (int i = tid * 4; i < 128 * 128; i += 256 * 4) {
    float4 v = *(const float4*)(src + i);
    xs[i + 0] = f2bf(v.x); xs[i + 1] = f2bf(v.y);
    xs[i + 2] = f2bf(v.z); xs[i + 3] = f2bf(v.w);
  }
  asm volatile("s_wait_asynccnt 0x0" ::: "memory");  // own DMAs landed
  __syncthreads();                                   // => all waves' DMAs landed

  const int wv = tid >> 5, lane = tid & 31;
  const int lg = lane >> 4, lr = lane & 15;
  const int m0 = wv * 16;

  float cacc[8][8];
#pragma unroll
  for (int nt = 0; nt < 8; ++nt)
#pragma unroll
    for (int r = 0; r < 8; ++r) cacc[nt][r] = 0.f;

  for (int ke = 0; ke < 2; ++ke) {
    const int e = (ke == 0) ? e0 : e1;
    const float g = gval[b * 2 + ke];
    const unsigned short* wl = smem + XS_USH + ke * WL_USH;   // LDS weight tile
    float ebv[8];
#pragma unroll
    for (int nt = 0; nt < 8; ++nt) ebv[nt] = expert_b[e * 128 + nt * 16 + lr];

    v8f zero = {};
    v8f acc[8];
#pragma unroll
    for (int nt = 0; nt < 8; ++nt) acc[nt] = zero;

    for (int kk = 0; kk < 4; ++kk) {         // K = 128 -> 4 steps of 32
      V16 a;
      const unsigned short* ap = &xs[(m0 + lr) * 128 + kk * 32 + lg * 8];
      a.q[0] = *(const uint4*)ap;
      a.q[1] = *(const uint4*)(ap + 16);
#pragma unroll
      for (int nt = 0; nt < 8; ++nt) {
        V16 bfr;
        const unsigned short* bp = wl + (nt * 16 + lr) * 128 + kk * 32 + lg * 8;
        bfr.q[0] = *(const uint4*)bp;
        bfr.q[1] = *(const uint4*)(bp + 16);
        acc[nt] = WMMA_BF16(a.v, bfr.v, acc[nt]);
      }
    }
#pragma unroll
    for (int nt = 0; nt < 8; ++nt)
#pragma unroll
      for (int r = 0; r < 8; ++r)
        cacc[nt][r] += g * __expf(acc[nt][r] + ebv[nt]);
  }

  // ---- store log(combined) ----
  float* orow = out + row0 * DD;
#pragma unroll
  for (int nt = 0; nt < 8; ++nt) {
    int c = nt * 16 + lr;
#pragma unroll
    for (int r = 0; r < 8; ++r) {
      int m = m0 + r + lg * 8;
      float v = cacc[nt][r];
      v = (v == 0.f) ? COMBINE_EPS : v;
      orow[(long)m * 128 + c] = __logf(v);
    }
  }
}

// ---------------------------------------------------------------------------
// Host launcher (graph-capture safe: no malloc/sync, ws-based scratch).
// ws usage ~8.4 MiB.
// ---------------------------------------------------------------------------
extern "C" void kernel_launch(void* const* d_in, const int* in_sizes, int n_in,
                              void* d_out, int out_size, void* d_ws, size_t ws_size,
                              hipStream_t stream) {
  (void)in_sizes; (void)n_in; (void)out_size; (void)ws_size;
  const float* x        = (const float*)d_in[0];
  const float* conv_w   = (const float*)d_in[1];
  const float* conv_b   = (const float*)d_in[2];
  const float* bn_gamma = (const float*)d_in[3];
  const float* bn_beta  = (const float*)d_in[4];
  const float* bn_mean  = (const float*)d_in[5];
  const float* bn_var   = (const float*)d_in[6];
  const float* fuse_w   = (const float*)d_in[7];
  const float* fuse_b   = (const float*)d_in[8];
  const float* w_gate   = (const float*)d_in[9];
  const float* expert_w = (const float*)d_in[10];
  const float* expert_b = (const float*)d_in[11];

  char* ws = (char*)d_ws;
  size_t off = 0;
  auto alloc = [&](size_t bytes) {
    size_t o = off;
    off = (off + bytes + 255) & ~(size_t)255;
    return o;
  };
  unsigned short* wT_conv = (unsigned short*)(ws + alloc(5 * 128 * CONV_K * 2));
  unsigned short* wT_exp  = (unsigned short*)(ws + alloc(NEXP * 128 * 128 * 2));
  float*          sb      = (float*)(ws + alloc(2 * 640 * 4));
  unsigned short* actA    = (unsigned short*)(ws + alloc((size_t)NIMG * 16 * 16 * 128 * 2));
  unsigned short* actB    = (unsigned short*)(ws + alloc((size_t)NIMG * 8 * 8 * 128 * 2));
  int*            gidx    = (int*)(ws + alloc(BB * 2 * 4));
  float*          gvals   = (float*)(ws + alloc(BB * 2 * 4));

  prep_kernel<<<256, 256, 0, stream>>>(conv_w, conv_b, bn_gamma, bn_beta, bn_mean,
                                       bn_var, expert_w, wT_conv, wT_exp, sb);

  // conv tower: (s_in, s_out) = (32,16)(16,8)(8,4)(4,2)(2,1); ping-pong A/B
  {
    int M = NIMG * 16 * 16;
    conv_kernel<true><<<(M + 63) / 64, 128, 65536, stream>>>(
        x, nullptr, wT_conv, sb, sb + 640, actA, 32, 16, M);
  }
  {
    int M = NIMG * 8 * 8;
    conv_kernel<false><<<(M + 63) / 64, 128, 65536, stream>>>(
        nullptr, actA, wT_conv + 1 * 128 * CONV_K, sb + 128, sb + 640 + 128, actB, 16, 8, M);
  }
  {
    int M = NIMG * 4 * 4;
    conv_kernel<false><<<(M + 63) / 64, 128, 65536, stream>>>(
        nullptr, actB, wT_conv + 2 * 128 * CONV_K, sb + 256, sb + 640 + 256, actA, 8, 4, M);
  }
  {
    int M = NIMG * 2 * 2;
    conv_kernel<false><<<(M + 63) / 64, 128, 65536, stream>>>(
        nullptr, actA, wT_conv + 3 * 128 * CONV_K, sb + 384, sb + 640 + 384, actB, 4, 2, M);
  }
  {
    int M = NIMG;
    conv_kernel<false><<<(M + 63) / 64, 128, 65536, stream>>>(
        nullptr, actB, wT_conv + 4 * 128 * CONV_K, sb + 512, sb + 640 + 512, actA, 2, 1, M);
  }

  gate_kernel<<<1, 128, 0, stream>>>(actA, fuse_w, fuse_b, w_gate, gidx, gvals);

  // 96KB dynamic LDS: x tile + two async-DMA'd expert weight tiles
  expert_kernel<<<TOKENS / 128, 256, 96 * 1024, stream>>>(x, wT_exp, expert_b,
                                                          gidx, gvals, (float*)d_out);
}
